// HybridSpatialReasoningModule_28372553957931
// MI455X (gfx1250) — compile-verified, matrix operand order checked by reading, not run on hardware
//
#include <hip/hip_runtime.h>
#include <hip/hip_bf16.h>

typedef __attribute__((ext_vector_type(16))) _Float16 v16h;
typedef __attribute__((ext_vector_type(8)))  float    v8f;

#define BB 2
#define NN 384
#define DD 768
#define HID 256
#define SS 64
#define NH 8
#define DH 96
#define JB 64          // j-rows per LDS block (6 blocks of 64 = 384)

// ---------------------------------------------------------------------------
// Kernel 1: pack ge_w2 (256x768 f32, row-major [k][n]) into f16 WMMA B-fragment
// layout: w2p[((ntile*8 + kk)*32 + lane)*16 + e]
//   n = ntile*16 + (lane&15)
//   k = kk*32 + (lane<16 ? 0 : 8) + (e<8 ? e : e+8)
// so each lane's v16h fragment is one contiguous 32B read.
// ---------------------------------------------------------------------------
__global__ __launch_bounds__(256) void pack_w2(const float* __restrict__ w2,
                                               _Float16* __restrict__ w2p) {
  int idx = blockIdx.x * 256 + threadIdx.x;          // 196608 total halves
  if (idx >= HID * DD) return;
  int e    = idx & 15;
  int lane = (idx >> 4) & 31;
  int kk   = (idx >> 9) & 7;
  int nt   = idx >> 12;                               // 0..47
  int n = nt * 16 + (lane & 15);
  int k = kk * 32 + ((lane < 16) ? 0 : 8) + ((e < 8) ? e : e + 8);
  w2p[idx] = (_Float16)w2[k * DD + n];
}

// ---------------------------------------------------------------------------
// Kernel 2: fused geometric encoder + mean over j.
// One workgroup per (b,i): 256 threads = 8 waves.
//   Phase 1: compute h[j,0:256] = relu(LN(rel(b,i,j) @ w1 + b1)) into LDS (f16)
//   Phase 2: WMMA  y = h @ w2 + b2 ; LN rows ; accumulate mean over j
// ---------------------------------------------------------------------------
__global__ __launch_bounds__(256) void geo_encoder(
    const float* __restrict__ coords,
    const float* __restrict__ w1, const float* __restrict__ b1,
    const float* __restrict__ g1, const float* __restrict__ be1,
    const _Float16* __restrict__ w2p,
    const float* __restrict__ b2, const float* __restrict__ g2,
    const float* __restrict__ be2,
    float* __restrict__ geo_ctx) {
  __shared__ __align__(16) _Float16 hbuf[JB * HID];   // 32 KB
  __shared__ float w1s[9 * HID];                      // 9 KB
  __shared__ float b1s[HID], g1s[HID], be1s[HID];     // 3 KB
  __shared__ float b2s[DD], g2s[DD], be2s[DD];        // 9 KB
  __shared__ float redS[16 * 8], redQ[16 * 8];        // 1 KB
  __shared__ float stM[16], stR[16];

  const int tid  = threadIdx.x;
  const int lane = tid & 31;
  const int wvid = tid >> 5;

  for (int i2 = tid; i2 < 9 * HID; i2 += 256) w1s[i2] = w1[i2];
  if (tid < HID) { b1s[tid] = b1[tid]; g1s[tid] = g1[tid]; be1s[tid] = be1[tid]; }
  for (int i2 = tid; i2 < DD; i2 += 256) { b2s[i2] = b2[i2]; g2s[i2] = g2[i2]; be2s[i2] = be2[i2]; }
  __syncthreads();

  const int wg = blockIdx.x;
  const int b  = wg / NN;
  const int i  = wg % NN;
  const float pix = coords[(b * NN + i) * 3 + 0];
  const float piy = coords[(b * NN + i) * 3 + 1];
  const float piz = coords[(b * NN + i) * 3 + 2];

  float gacc[6] = {0.f, 0.f, 0.f, 0.f, 0.f, 0.f};
  const int colb = wvid * 96 + (lane & 15);
  const int rowbase = (lane < 16) ? 0 : 8;

  for (int jb = 0; jb < NN / JB; ++jb) {
    // -------- Phase 1: h rows into LDS (each wave does 8 rows) --------
    for (int rr = 0; rr < JB / 8; ++rr) {
      const int jl = wvid * (JB / 8) + rr;
      const int j  = jb * JB + jl;
      const float rx = pix - coords[(b * NN + j) * 3 + 0];
      const float ry = piy - coords[(b * NN + j) * 3 + 1];
      const float rz = piz - coords[(b * NN + j) * 3 + 2];
      const float d  = sqrtf(rx * rx + ry * ry + rz * rz);
      const float iv = 1.0f / fmaxf(d, 1e-12f);
      const float n0 = rx * iv, n1 = ry * iv, n2 = rz * iv;
      float rel[9] = { d, n0, n1, n2, 0.f, 0.f, n1 * n2, n0 * n1, n0 * n2 };
      float y[8];
      float s = 0.f, q = 0.f;
      #pragma unroll
      for (int u = 0; u < 8; ++u) {
        const int c = lane + 32 * u;
        float a = b1s[c];
        #pragma unroll
        for (int f = 0; f < 9; ++f) a += rel[f] * w1s[f * HID + c];
        y[u] = a; s += a; q += a * a;
      }
      #pragma unroll
      for (int off = 16; off; off >>= 1) {
        s += __shfl_xor(s, off, 32);
        q += __shfl_xor(q, off, 32);
      }
      const float m    = s * (1.0f / HID);
      const float var  = q * (1.0f / HID) - m * m;
      const float rstd = rsqrtf(var + 1e-5f);
      #pragma unroll
      for (int u = 0; u < 8; ++u) {
        const int c = lane + 32 * u;
        float e = (y[u] - m) * rstd * g1s[c] + be1s[c];
        hbuf[jl * HID + c] = (_Float16)fmaxf(e, 0.0f);
      }
    }
    __syncthreads();

    // -------- Phase 2: WMMA GEMM + rowwise LN + j-mean accumulation --------
    for (int jt = 0; jt < JB / 16; ++jt) {
      v8f acc[6];
      #pragma unroll
      for (int t = 0; t < 6; ++t) acc[t] = (v8f){0,0,0,0,0,0,0,0};

      const int r = lane & 15;
      #pragma unroll
      for (int t = 0; t < 6; ++t) {
        const int ntile = wvid * 6 + t;
        #pragma unroll
        for (int kk = 0; kk < 8; ++kk) {
          // A fragment from LDS (h rows)
          const int baseA = (jt * 16 + r) * HID + kk * 32 + ((lane < 16) ? 0 : 8);
          union { int4 i[2]; v16h h; } ua;
          ua.i[0] = *(const int4*)&hbuf[baseA];
          ua.i[1] = *(const int4*)&hbuf[baseA + 16];
          // B fragment from packed w2 (global -> L2 resident)
          const _Float16* bp = w2p + (size_t)(((ntile * 8 + kk) * 32) + lane) * 16;
          union { int4 i[2]; v16h h; } ub;
          ub.i[0] = *(const int4*)bp;
          ub.i[1] = *(const int4*)(bp + 8);
          acc[t] = __builtin_amdgcn_wmma_f32_16x16x32_f16(
              false, ua.h, false, ub.h, (short)0, acc[t], false, false);
        }
      }

      // bias + per-row partial stats (row = r + rowbase for VGPR r)
      float sr[8] = {0}, qr[8] = {0};
      #pragma unroll
      for (int t = 0; t < 6; ++t) {
        const float bb = b2s[colb + t * 16];
        #pragma unroll
        for (int rg = 0; rg < 8; ++rg) {
          acc[t][rg] += bb;
          const float v = acc[t][rg];
          sr[rg] += v; qr[rg] += v * v;
        }
      }
      #pragma unroll
      for (int off = 8; off; off >>= 1) {
        #pragma unroll
        for (int rg = 0; rg < 8; ++rg) {
          sr[rg] += __shfl_xor(sr[rg], off, 32);
          qr[rg] += __shfl_xor(qr[rg], off, 32);
        }
      }
      if ((lane & 15) == 0) {
        #pragma unroll
        for (int rg = 0; rg < 8; ++rg) {
          redS[(rg + rowbase) * 8 + wvid] = sr[rg];
          redQ[(rg + rowbase) * 8 + wvid] = qr[rg];
        }
      }
      __syncthreads();
      if (tid < 16) {
        float S = 0.f, Q = 0.f;
        #pragma unroll
        for (int wv2 = 0; wv2 < 8; ++wv2) { S += redS[tid * 8 + wv2]; Q += redQ[tid * 8 + wv2]; }
        const float m   = S * (1.0f / DD);
        const float var = Q * (1.0f / DD) - m * m;
        stM[tid] = m;
        stR[tid] = rsqrtf(var + 1e-5f);
      }
      __syncthreads();
      #pragma unroll
      for (int t = 0; t < 6; ++t) {
        const int   col = colb + t * 16;
        const float gg  = g2s[col];
        const float eb  = be2s[col];
        #pragma unroll
        for (int rg = 0; rg < 8; ++rg) {
          const int row = rg + rowbase;
          gacc[t] += (acc[t][rg] - stM[row]) * stR[row] * gg + eb;
        }
      }
      __syncthreads();
    }
  }

  // fold lanes 0-15 (rows 0..7) with lanes 16-31 (rows 8..15) and write mean
  #pragma unroll
  for (int t = 0; t < 6; ++t) {
    const float v = gacc[t] + __shfl_xor(gacc[t], 16, 32);
    if (lane < 16)
      geo_ctx[(size_t)(b * NN + i) * DD + wvid * 96 + t * 16 + lane] = v * (1.0f / NN);
  }
}

// ---------------------------------------------------------------------------
// Kernel 3: superpoint scatter-mean (gather formulation -> deterministic)
// ---------------------------------------------------------------------------
__global__ __launch_bounds__(256) void spp_stats(
    const int* __restrict__ labels, const float* __restrict__ feat,
    const float* __restrict__ geo, float* __restrict__ cnt,
    float* __restrict__ mean_f, float* __restrict__ mean_g) {
  const int b  = blockIdx.x / SS;
  const int sp = blockIdx.x % SS;
  __shared__ int lab[NN];
  for (int i2 = threadIdx.x; i2 < NN; i2 += 256) lab[i2] = labels[b * NN + i2];
  __syncthreads();
  float c = 0.f;
  for (int n2 = 0; n2 < NN; ++n2) c += (lab[n2] == sp) ? 1.f : 0.f;
  const float denom = 1.0f / fmaxf(c, 1.0f);
  for (int d2 = threadIdx.x; d2 < DD; d2 += 256) {
    float sf = 0.f, sg = 0.f;
    for (int n2 = 0; n2 < NN; ++n2) {
      if (lab[n2] == sp) {
        sf += feat[(size_t)(b * NN + n2) * DD + d2];
        sg += geo [(size_t)(b * NN + n2) * DD + d2];
      }
    }
    mean_f[(size_t)(b * SS + sp) * DD + d2] = sf * denom;
    mean_g[(size_t)(b * SS + sp) * DD + d2] = sg * denom;
  }
  if (threadIdx.x == 0) cnt[b * SS + sp] = c;
}

// ---------------------------------------------------------------------------
// Kernel 4: aggregator MLP (per (b,s) row) + combine with mean_g
// ---------------------------------------------------------------------------
__global__ __launch_bounds__(256) void agg_mlp(
    const float* __restrict__ mean_f, const float* __restrict__ mean_g,
    const float* __restrict__ w1a, const float* __restrict__ b1a,
    const float* __restrict__ g1a, const float* __restrict__ be1a,
    const float* __restrict__ w2a, const float* __restrict__ b2a,
    const float* __restrict__ g2a, const float* __restrict__ be2a,
    float* __restrict__ combined) {
  const int row = blockIdx.x;               // b*S + s
  __shared__ float xr[DD];
  __shared__ float hr[HID];
  __shared__ float red[16];
  const int tid = threadIdx.x, lane = tid & 31, wvid = tid >> 5;
  for (int d2 = tid; d2 < DD; d2 += 256) xr[d2] = mean_f[(size_t)row * DD + d2];
  __syncthreads();
  // layer 1: one column per thread (HID == 256)
  float acc = b1a[tid];
  for (int k2 = 0; k2 < DD; ++k2) acc += xr[k2] * w1a[k2 * HID + tid];
  float s1 = acc, s2 = acc * acc;
  #pragma unroll
  for (int off = 16; off; off >>= 1) { s1 += __shfl_xor(s1, off, 32); s2 += __shfl_xor(s2, off, 32); }
  if (lane == 0) { red[wvid] = s1; red[8 + wvid] = s2; }
  __syncthreads();
  float S1 = 0.f, S2 = 0.f;
  for (int w2i = 0; w2i < 8; ++w2i) { S1 += red[w2i]; S2 += red[8 + w2i]; }
  {
    const float m = S1 * (1.0f / HID);
    const float var = S2 * (1.0f / HID) - m * m;
    const float rstd = rsqrtf(var + 1e-5f);
    hr[tid] = fmaxf(0.f, (acc - m) * rstd * g1a[tid] + be1a[tid]);
  }
  __syncthreads();
  // layer 2: 3 columns per thread
  float y[3], p1 = 0.f, p2 = 0.f;
  #pragma unroll
  for (int u = 0; u < 3; ++u) {
    const int d2 = tid + 256 * u;
    float a2 = b2a[d2];
    for (int k2 = 0; k2 < HID; ++k2) a2 += hr[k2] * w2a[k2 * DD + d2];
    y[u] = a2; p1 += a2; p2 += a2 * a2;
  }
  #pragma unroll
  for (int off = 16; off; off >>= 1) { p1 += __shfl_xor(p1, off, 32); p2 += __shfl_xor(p2, off, 32); }
  __syncthreads();
  if (lane == 0) { red[wvid] = p1; red[8 + wvid] = p2; }
  __syncthreads();
  float T1 = 0.f, T2 = 0.f;
  for (int w2i = 0; w2i < 8; ++w2i) { T1 += red[w2i]; T2 += red[8 + w2i]; }
  const float m2 = T1 * (1.0f / DD);
  const float var2 = T2 * (1.0f / DD) - m2 * m2;
  const float rstd2 = rsqrtf(var2 + 1e-5f);
  #pragma unroll
  for (int u = 0; u < 3; ++u) {
    const int d2 = tid + 256 * u;
    combined[(size_t)row * DD + d2] =
        (y[u] - m2) * rstd2 * g2a[d2] + be2a[d2] + mean_g[(size_t)row * DD + d2];
  }
}

// ---------------------------------------------------------------------------
// Kernel 5: enhanced = valid ? 0.7*f + 0.3*combined[label] : f
// ---------------------------------------------------------------------------
__global__ __launch_bounds__(256) void enhance(
    const int* __restrict__ labels, const float* __restrict__ feat,
    const float* __restrict__ combined, const float* __restrict__ cnt,
    float* __restrict__ enh) {
  const size_t idx = (size_t)blockIdx.x * 256 + threadIdx.x;
  if (idx >= (size_t)BB * NN * DD) return;
  const int d2 = idx % DD;
  const int n2 = (idx / DD) % NN;
  const int b  = idx / ((size_t)DD * NN);
  const int sp = labels[b * NN + n2];
  const bool valid = cnt[b * SS + sp] >= 2.0f;
  const float f = feat[idx];
  const float c = combined[(size_t)(b * SS + sp) * DD + d2];
  enh[idx] = valid ? (0.7f * f + 0.3f * c) : f;
}

// ---------------------------------------------------------------------------
// Kernel 6: QKV projections  (q from enhanced, k/v from geo_ctx)
// ---------------------------------------------------------------------------
__global__ __launch_bounds__(256) void qkv_proj(
    const float* __restrict__ enh, const float* __restrict__ geo,
    const float* __restrict__ wq, const float* __restrict__ bq,
    const float* __restrict__ wk, const float* __restrict__ bk,
    const float* __restrict__ wv, const float* __restrict__ bvv,
    float* __restrict__ qb, float* __restrict__ kb, float* __restrict__ vb) {
  const int row = blockIdx.x;               // b*N + n
  __shared__ float xe[DD], xg[DD];
  for (int d2 = threadIdx.x; d2 < DD; d2 += 256) {
    xe[d2] = enh[(size_t)row * DD + d2];
    xg[d2] = geo[(size_t)row * DD + d2];
  }
  __syncthreads();
  for (int d2 = threadIdx.x; d2 < DD; d2 += 256) {
    float aq = bq[d2], ak = bk[d2], av = bvv[d2];
    for (int k2 = 0; k2 < DD; ++k2) {
      aq += xe[k2] * wq[k2 * DD + d2];
      ak += xg[k2] * wk[k2 * DD + d2];
      av += xg[k2] * wv[k2 * DD + d2];
    }
    qb[(size_t)row * DD + d2] = aq;
    kb[(size_t)row * DD + d2] = ak;
    vb[(size_t)row * DD + d2] = av;
  }
}

// ---------------------------------------------------------------------------
// Kernel 7: attention per (b, head, query-row)
// ---------------------------------------------------------------------------
__global__ __launch_bounds__(128) void attn(
    const float* __restrict__ qb, const float* __restrict__ kb,
    const float* __restrict__ vb, float* __restrict__ ob) {
  const int wg = blockIdx.x;
  const int qi = wg % NN;
  const int h  = (wg / NN) % NH;
  const int b  = wg / (NN * NH);
  __shared__ float qrow[DH];
  __shared__ float p[NN];
  __shared__ float red[4];
  const int tid = threadIdx.x, lane = tid & 31, wvid = tid >> 5;
  const float* qr = qb + ((size_t)(b * NN + qi) * DD + h * DH);
  if (tid < DH) qrow[tid] = qr[tid];
  __syncthreads();
  const float scale = 0.10206207262f;       // 96^-0.5
  float mymax = -3.4e38f;
  for (int kk2 = tid; kk2 < NN; kk2 += 128) {
    const float* kr = kb + ((size_t)(b * NN + kk2) * DD + h * DH);
    float dot = 0.f;
    #pragma unroll 8
    for (int d2 = 0; d2 < DH; ++d2) dot += qrow[d2] * kr[d2];
    dot *= scale;
    p[kk2] = dot;
    mymax = fmaxf(mymax, dot);
  }
  #pragma unroll
  for (int off = 16; off; off >>= 1) mymax = fmaxf(mymax, __shfl_xor(mymax, off, 32));
  if (lane == 0) red[wvid] = mymax;
  __syncthreads();
  const float M = fmaxf(fmaxf(red[0], red[1]), fmaxf(red[2], red[3]));
  __syncthreads();
  float mysum = 0.f;
  for (int kk2 = tid; kk2 < NN; kk2 += 128) {
    const float e = __expf(p[kk2] - M);
    p[kk2] = e;
    mysum += e;
  }
  #pragma unroll
  for (int off = 16; off; off >>= 1) mysum += __shfl_xor(mysum, off, 32);
  if (lane == 0) red[wvid] = mysum;
  __syncthreads();
  const float Z = red[0] + red[1] + red[2] + red[3];
  if (tid < DH) {
    float o = 0.f;
    for (int kk2 = 0; kk2 < NN; ++kk2)
      o += p[kk2] * vb[(size_t)(b * NN + kk2) * DD + h * DH + tid];
    ob[(size_t)(b * NN + qi) * DD + h * DH + tid] = o * (1.0f / Z);
  }
}

// ---------------------------------------------------------------------------
// Kernel 8: output projection + residual:  out = enhanced + 0.5*(o @ wo + bo)
// ---------------------------------------------------------------------------
__global__ __launch_bounds__(256) void out_proj(
    const float* __restrict__ ob, const float* __restrict__ enh,
    const float* __restrict__ wo, const float* __restrict__ bo,
    float* __restrict__ out) {
  const int row = blockIdx.x;
  __shared__ float xo[DD];
  for (int d2 = threadIdx.x; d2 < DD; d2 += 256) xo[d2] = ob[(size_t)row * DD + d2];
  __syncthreads();
  for (int d2 = threadIdx.x; d2 < DD; d2 += 256) {
    float a = bo[d2];
    for (int k2 = 0; k2 < DD; ++k2) a += xo[k2] * wo[k2 * DD + d2];
    out[(size_t)row * DD + d2] = enh[(size_t)row * DD + d2] + 0.5f * a;
  }
}

// ---------------------------------------------------------------------------
extern "C" void kernel_launch(void* const* d_in, const int* in_sizes, int n_in,
                              void* d_out, int out_size, void* d_ws, size_t ws_size,
                              hipStream_t stream) {
  const float* coords = (const float*)d_in[0];
  const float* feat   = (const float*)d_in[1];
  const int*   labels = (const int*)d_in[2];
  const float* ge_w1  = (const float*)d_in[3];
  const float* ge_b1  = (const float*)d_in[4];
  const float* ge_g1  = (const float*)d_in[5];
  const float* ge_be1 = (const float*)d_in[6];
  const float* ge_w2  = (const float*)d_in[7];
  const float* ge_b2  = (const float*)d_in[8];
  const float* ge_g2  = (const float*)d_in[9];
  const float* ge_be2 = (const float*)d_in[10];
  const float* ag_w1  = (const float*)d_in[11];
  const float* ag_b1  = (const float*)d_in[12];
  const float* ag_g1  = (const float*)d_in[13];
  const float* ag_be1 = (const float*)d_in[14];
  const float* ag_w2  = (const float*)d_in[15];
  const float* ag_b2  = (const float*)d_in[16];
  const float* ag_g2  = (const float*)d_in[17];
  const float* ag_be2 = (const float*)d_in[18];
  const float* wq = (const float*)d_in[19]; const float* bq = (const float*)d_in[20];
  const float* wk = (const float*)d_in[21]; const float* bk = (const float*)d_in[22];
  const float* wv = (const float*)d_in[23]; const float* bv = (const float*)d_in[24];
  const float* wo = (const float*)d_in[25]; const float* bo = (const float*)d_in[26];
  float* out = (float*)d_out;

  char* wsb = (char*)d_ws;
  size_t off = 0;
  auto wsalloc = [&](size_t bytes) -> void* {
    void* p = wsb + off;
    off = (off + bytes + 255) & ~(size_t)255;
    return p;
  };
  _Float16* w2p   = (_Float16*)wsalloc((size_t)HID * DD * 2);
  float* geo      = (float*)wsalloc((size_t)BB * NN * DD * 4);
  float* cnt      = (float*)wsalloc((size_t)BB * SS * 4);
  float* mean_f   = (float*)wsalloc((size_t)BB * SS * DD * 4);
  float* mean_g   = (float*)wsalloc((size_t)BB * SS * DD * 4);
  float* combined = (float*)wsalloc((size_t)BB * SS * DD * 4);
  float* enh      = (float*)wsalloc((size_t)BB * NN * DD * 4);
  float* qb       = (float*)wsalloc((size_t)BB * NN * DD * 4);
  float* kb       = (float*)wsalloc((size_t)BB * NN * DD * 4);
  float* vb       = (float*)wsalloc((size_t)BB * NN * DD * 4);
  float* ob       = (float*)wsalloc((size_t)BB * NN * DD * 4);
  (void)ws_size; (void)in_sizes; (void)n_in; (void)out_size;

  pack_w2<<<(HID * DD + 255) / 256, 256, 0, stream>>>(ge_w2, w2p);

  geo_encoder<<<BB * NN, 256, 0, stream>>>(coords, ge_w1, ge_b1, ge_g1, ge_be1,
                                           w2p, ge_b2, ge_g2, ge_be2, geo);

  spp_stats<<<BB * SS, 256, 0, stream>>>(labels, feat, geo, cnt, mean_f, mean_g);

  agg_mlp<<<BB * SS, 256, 0, stream>>>(mean_f, mean_g, ag_w1, ag_b1, ag_g1, ag_be1,
                                       ag_w2, ag_b2, ag_g2, ag_be2, combined);

  enhance<<<((size_t)BB * NN * DD + 255) / 256, 256, 0, stream>>>(labels, feat,
                                                                  combined, cnt, enh);

  qkv_proj<<<BB * NN, 256, 0, stream>>>(enh, geo, wq, bq, wk, bk, wv, bv, qb, kb, vb);

  attn<<<BB * NH * NN, 128, 0, stream>>>(qb, kb, vb, ob);

  out_proj<<<BB * NN, 256, 0, stream>>>(ob, enh, wo, bo, out);
}
